// MemoryEfficientAttention_27960237097287
// MI455X (gfx1250) — compile-verified, matrix operand order checked by reading, not run on hardware
//
#include <hip/hip_runtime.h>
#include <hip/hip_bf16.h>

typedef __attribute__((ext_vector_type(16))) _Float16 v16h;
typedef __attribute__((ext_vector_type(8)))  _Float16 v8h;
typedef __attribute__((ext_vector_type(4)))  _Float16 v4h;
typedef __attribute__((ext_vector_type(8)))  float    v8f;

#define EMBED  1024
#define NHEADS 16
#define HDIM   64
#define BS     8
#define TLEN   1024
#define MTOT   (BS * TLEN)      // 8192 rows for all big GEMMs
#define NEGBIG (-1.0e30f)

// ---------------------------------------------------------------------------
// Load a 16x32 f16 fragment in the CDNA5 WMMA A-layout from row-major data.
// Lane L (0..15) holds row M=L, lanes 16..31 hold the +8 K-offset half.
// VGPR0..3 = K {k0..k0+7 | k0+8..k0+15}, VGPR4..7 = K {k0+16.. | k0+24..}.
// Works for both A and for B when given B^T (row-major [N,K]) data.
// ---------------------------------------------------------------------------
__device__ __forceinline__ v16h load_frag16x32(const _Float16* base, int ld,
                                               int lane, int k0) {
  const _Float16* p = base + (size_t)(lane & 15) * ld + (k0 + ((lane >> 4) << 3));
  v8h lo = *(const v8h*)p;          // K = koff .. koff+7
  v8h hi = *(const v8h*)(p + 16);   // K = koff+16 .. koff+23
  return __builtin_shufflevector(lo, hi, 0, 1, 2, 3, 4, 5, 6, 7,
                                         8, 9, 10, 11, 12, 13, 14, 15);
}

__device__ __forceinline__ v8f wmma_f16(v16h a, v16h b, v8f c) {
  return __builtin_amdgcn_wmma_f32_16x16x32_f16(false, a, false, b,
                                                (short)0, c, false, false);
}

// ---------------------------------------------------------------------------
// fp32 -> f16 conversion, 4 elements / lane
// ---------------------------------------------------------------------------
__global__ __launch_bounds__(256)
void cvt_f32_to_f16(const float* __restrict__ in, _Float16* __restrict__ out, int n) {
  int i = (blockIdx.x * blockDim.x + threadIdx.x) * 4;
  if (i + 3 < n) {
    float4 f = *(const float4*)(in + i);
    v4h h;
    h[0] = (_Float16)f.x; h[1] = (_Float16)f.y;
    h[2] = (_Float16)f.z; h[3] = (_Float16)f.w;
    *(v4h*)(out + i) = h;
  }
}

// ---------------------------------------------------------------------------
// GEMM: y[m,n] = sum_k A[m,k] * W[n,k] + bias[n]   (i.e. x @ W^T + b)
// A: [M,K] f16 row-major.  W: [N,K] f16 row-major (acts as B^T).
// Block = 256 threads = 8 waves; wave computes a 16x64 tile.
// MODE 0: f16 out, per-head layout  [b,h,t,d]       (Q, K)
// MODE 1: f16 out, transposed head  [b,h,d,t]       (V^T)
// MODE 2: f32 out, plain row-major  [m,n]           (final projection)
// ---------------------------------------------------------------------------
template <int MODE>
__global__ __launch_bounds__(256)
void gemm_f16_wmma(const _Float16* __restrict__ A, const _Float16* __restrict__ W,
                   const float* __restrict__ bias, void* __restrict__ out,
                   int M, int N, int K) {
  const int lane = threadIdx.x & 31;
  const int wv   = threadIdx.x >> 5;
  const int m0   = blockIdx.y * 128 + wv * 16;
  const int n0   = blockIdx.x * 64;

  v8f acc[4] = {};
  const _Float16* Arow = A + (size_t)m0 * K;

  for (int k0 = 0; k0 < K; k0 += 32) {
    v16h a = load_frag16x32(Arow, K, lane, k0);
#pragma unroll
    for (int nt = 0; nt < 4; ++nt) {
      v16h b = load_frag16x32(W + (size_t)(n0 + nt * 16) * K, K, lane, k0);
      acc[nt] = wmma_f16(a, b, acc[nt]);
    }
  }

  // C layout: element acc[nt][r] lives at row M = r + 8*(lane>=16), col = lane&15
  const int mbase = m0 + 8 * (lane >> 4);
#pragma unroll
  for (int nt = 0; nt < 4; ++nt) {
    const int col = n0 + nt * 16 + (lane & 15);
    const float bval = bias[col];
#pragma unroll
    for (int r = 0; r < 8; ++r) {
      const int m = mbase + r;
      const float val = acc[nt][r] + bval;
      if (MODE == 2) {
        ((float*)out)[(size_t)m * N + col] = val;
      } else {
        const int b = m >> 10, t = m & 1023;      // m = b*TLEN + t
        const int h = col >> 6, d = col & 63;     // col = h*HDIM + d
        _Float16* o16 = (_Float16*)out;
        if (MODE == 0)
          o16[(((size_t)(b * NHEADS + h) * TLEN) + t) * HDIM + d] = (_Float16)val;
        else
          o16[(((size_t)(b * NHEADS + h) * HDIM) + d) * TLEN + t] = (_Float16)val;
      }
    }
  }
}

// ---------------------------------------------------------------------------
// Flash attention with the chunked mask (CHUNK == 64 == tile size, so the
// mask is just "k > q inside the diagonal key-tile").
// Block = 128 threads = 4 waves; one 64-query chunk per block, wave owns
// 16 query rows.  Online softmax, P staged through LDS (C-layout->A-layout).
// ---------------------------------------------------------------------------
__global__ __launch_bounds__(128)
void attn_fwd(const _Float16* __restrict__ qh, const _Float16* __restrict__ kh,
              const _Float16* __restrict__ vT, const float* __restrict__ scale_p,
              _Float16* __restrict__ attn) {
  const int lane  = threadIdx.x & 31;
  const int wv    = threadIdx.x >> 5;
  const int qtile = blockIdx.x;          // 64-row query chunk == mask chunk
  const int bh    = blockIdx.y;          // b*NHEADS + h
  const int qrow0 = qtile * 64 + wv * 16;
  const int ncol  = lane & 15;
  const int rbase = 8 * (lane >> 4);     // row offset of this lane's C elements

  const _Float16* Q  = qh + (size_t)bh * TLEN * HDIM;
  const _Float16* Km = kh + (size_t)bh * TLEN * HDIM;
  const _Float16* Vm = vT + (size_t)bh * HDIM * TLEN;
  const float scale = scale_p[0];

  __shared__ __align__(16) _Float16 pshare[4][16 * 80];  // stride 80 keeps 16B align
  _Float16* pw = &pshare[wv][0];

  const v16h qf0 = load_frag16x32(Q + (size_t)qrow0 * HDIM, HDIM, lane, 0);
  const v16h qf1 = load_frag16x32(Q + (size_t)qrow0 * HDIM, HDIM, lane, 32);

  v8f o[4] = {};
  float mrow[8], lrow[8];
#pragma unroll
  for (int r = 0; r < 8; ++r) { mrow[r] = NEGBIG; lrow[r] = 0.0f; }

  for (int j = 0; j < TLEN / 64; ++j) {
    // ---- S = scale * Q Kt  (16x64 tile in 4 C fragments) -----------------
    v8f s[4] = {};
#pragma unroll
    for (int nt = 0; nt < 4; ++nt) {
      v16h b0 = load_frag16x32(Km + (size_t)(j * 64 + nt * 16) * HDIM, HDIM, lane, 0);
      s[nt] = wmma_f16(qf0, b0, s[nt]);
      v16h b1 = load_frag16x32(Km + (size_t)(j * 64 + nt * 16) * HDIM, HDIM, lane, 32);
      s[nt] = wmma_f16(qf1, b1, s[nt]);
    }

    const bool diag = (j == qtile);
    float mnew[8];
#pragma unroll
    for (int r = 0; r < 8; ++r) {
      const int qg = qrow0 + rbase + r;
      float mx = NEGBIG;
#pragma unroll
      for (int nt = 0; nt < 4; ++nt) {
        float v = s[nt][r] * scale;
        const int kg = j * 64 + nt * 16 + ncol;
        if (diag && kg > qg) v = NEGBIG;     // masked -> exp() == 0
        s[nt][r] = v;
        mx = fmaxf(mx, v);
      }
      for (int d = 1; d < 16; d <<= 1) mx = fmaxf(mx, __shfl_xor(mx, d, 32));
      mnew[r] = fmaxf(mrow[r], mx);
    }

    // ---- online softmax update; stage P into LDS -------------------------
#pragma unroll
    for (int r = 0; r < 8; ++r) {
      const float alpha = __expf(mrow[r] - mnew[r]);
      float rs = 0.0f;
#pragma unroll
      for (int nt = 0; nt < 4; ++nt) {
        const float p = __expf(s[nt][r] - mnew[r]);
        s[nt][r] = p;
        rs += p;
      }
      for (int d = 1; d < 16; d <<= 1) rs += __shfl_xor(rs, d, 32);
      lrow[r] = lrow[r] * alpha + rs;
      mrow[r] = mnew[r];
      const int row = rbase + r;
#pragma unroll
      for (int nt = 0; nt < 4; ++nt) {
        o[nt][r] *= alpha;
        pw[row * 80 + nt * 16 + ncol] = (_Float16)s[nt][r];
      }
    }
    asm volatile("s_wait_dscnt 0" ::: "memory");  // same-wave LDS RAW

    const v16h pf0 = load_frag16x32(pw, 80, lane, 0);
    const v16h pf1 = load_frag16x32(pw, 80, lane, 32);

    // ---- O += P V   (V^T is row-major [d, t] -> B^T loader) --------------
#pragma unroll
    for (int nt = 0; nt < 4; ++nt) {
      v16h vb0 = load_frag16x32(Vm + (size_t)(nt * 16) * TLEN, TLEN, lane, j * 64);
      o[nt] = wmma_f16(pf0, vb0, o[nt]);
      v16h vb1 = load_frag16x32(Vm + (size_t)(nt * 16) * TLEN, TLEN, lane, j * 64 + 32);
      o[nt] = wmma_f16(pf1, vb1, o[nt]);
    }
  }

  // ---- epilogue: merge heads into [b, t, h*HDIM + d] f16 ------------------
  const int b = bh >> 4, h = bh & 15;
#pragma unroll
  for (int nt = 0; nt < 4; ++nt) {
    const int d = nt * 16 + ncol;
#pragma unroll
    for (int r = 0; r < 8; ++r) {
      const int t = qrow0 + rbase + r;
      attn[((size_t)(b * TLEN + t)) * EMBED + h * HDIM + d] =
          (_Float16)(o[nt][r] / lrow[r]);
    }
  }
}

// ---------------------------------------------------------------------------
extern "C" void kernel_launch(void* const* d_in, const int* in_sizes, int n_in,
                              void* d_out, int out_size, void* d_ws, size_t ws_size,
                              hipStream_t stream) {
  const float* x     = (const float*)d_in[0];
  const float* Wq    = (const float*)d_in[1];
  const float* bq    = (const float*)d_in[2];
  const float* Wk    = (const float*)d_in[3];
  const float* bk    = (const float*)d_in[4];
  const float* Wv    = (const float*)d_in[5];
  const float* bv    = (const float*)d_in[6];
  const float* Wo    = (const float*)d_in[7];
  const float* bo    = (const float*)d_in[8];
  const float* scale = (const float*)d_in[9];

  const size_t nBig = (size_t)MTOT * EMBED;    // 8M elements
  const size_t nW   = (size_t)EMBED * EMBED;   // 1M elements

  char* p = (char*)d_ws;
  _Float16* xh   = (_Float16*)p; p += nBig * 2;
  _Float16* wqh  = (_Float16*)p; p += nW * 2;
  _Float16* wkh  = (_Float16*)p; p += nW * 2;
  _Float16* wvh  = (_Float16*)p; p += nW * 2;
  _Float16* woh  = (_Float16*)p; p += nW * 2;
  _Float16* q16  = (_Float16*)p; p += nBig * 2;
  _Float16* k16  = (_Float16*)p; p += nBig * 2;
  _Float16* v16t = (_Float16*)p; p += nBig * 2;
  _Float16* at16 = (_Float16*)p; p += nBig * 2;

  cvt_f32_to_f16<<<(int)(nBig / 1024), 256, 0, stream>>>(x, xh, (int)nBig);
  cvt_f32_to_f16<<<(int)(nW / 1024), 256, 0, stream>>>(Wq, wqh, (int)nW);
  cvt_f32_to_f16<<<(int)(nW / 1024), 256, 0, stream>>>(Wk, wkh, (int)nW);
  cvt_f32_to_f16<<<(int)(nW / 1024), 256, 0, stream>>>(Wv, wvh, (int)nW);
  cvt_f32_to_f16<<<(int)(nW / 1024), 256, 0, stream>>>(Wo, woh, (int)nW);

  dim3 gg(EMBED / 64, MTOT / 128);
  gemm_f16_wmma<0><<<gg, 256, 0, stream>>>(xh, wqh, bq, q16, MTOT, EMBED, EMBED);
  gemm_f16_wmma<0><<<gg, 256, 0, stream>>>(xh, wkh, bk, k16, MTOT, EMBED, EMBED);
  gemm_f16_wmma<1><<<gg, 256, 0, stream>>>(xh, wvh, bv, v16t, MTOT, EMBED, EMBED);

  attn_fwd<<<dim3(TLEN / 64, BS * NHEADS), 128, 0, stream>>>(q16, k16, v16t, scale, at16);

  gemm_f16_wmma<2><<<gg, 256, 0, stream>>>(at16, woh, bo, d_out, MTOT, EMBED, EMBED);
}
